// hisan_28750511080082
// MI455X (gfx1250) — matrix operand order repaired and checked by reading, not compiled, not run on hardware
//
#include <hip/hip_runtime.h>
#include <hip/hip_bf16.h>

// ---------------- problem constants ----------------
constexpr int B_  = 32;     // docs
constexpr int S_  = 64;     // lines per doc
constexpr int W_  = 48;     // words per line
constexpr int E_  = 300;    // embed dim
constexpr int EP  = 320;    // embed dim padded to multiple of 32 (WMMA K)
constexpr int A_  = 512;    // attention dim
constexpr int C0_ = 10;
constexpr int C1_ = 25;
constexpr float SCALE_ = 0.04419417382415922f;   // 1/sqrt(512)

// LDS leading dimensions, padded for conflict-free banks (all 16B multiples):
constexpr int LDX = 328;    // x buffer rows (E-padded), 656B -> bank stride 36
constexpr int LDA = 520;    // A=512 rows, 1040B -> bank stride 4
constexpr int LDV = 72;     // Vt / P rows (K dim <= 64), 144B -> bank stride 36

typedef __attribute__((ext_vector_type(16))) _Float16 v16h;
typedef __attribute__((ext_vector_type(8)))  _Float16 v8h;
typedef __attribute__((ext_vector_type(8)))  float    v8f;

// ---------------- WMMA operand loaders (CDNA5 16x16x32 f16 layouts) ----------
// A-matrix 16x32: lane r (0..15) = row r; lanes 0-15 hold K {0-7,16-23},
// lanes 16-31 hold K {8-15,24-31} (ISA 7.12.2). Source row-major, ld halves.
static __device__ __forceinline__ v16h ld_tileA(const _Float16* base, int ld, int lane) {
  const int r  = lane & 15;
  const int kh = (lane >> 4) & 1;
  const _Float16* p = base + r * ld + kh * 8;
  v8h lo = *(const v8h*)(p);        // K = kh*8 + 0..7
  v8h hi = *(const v8h*)(p + 16);   // K = 16 + kh*8 + 0..7
  v16h out;
#pragma unroll
  for (int i = 0; i < 8; ++i) { out[i] = lo[i]; out[8 + i] = hi[i]; }
  return out;
}

// B-matrix 32x16: lane n = column n; lanes 0-15 hold K {0-7,16-23}, lanes 16-31 hold
// K {8-15,24-31} (same K order as A). Column n of B is a contiguous row of
// length-K data at base + n*ld (B stored as N x K).
static __device__ __forceinline__ v16h ld_tileB(const _Float16* base, int ld, int lane) {
  const int n  = lane & 15;
  const int kh = (lane >> 4) & 1;
  const _Float16* p = base + n * ld + kh * 8;
  v8h lo = *(const v8h*)(p);
  v8h hi = *(const v8h*)(p + 16);
  v16h out;
#pragma unroll
  for (int i = 0; i < 8; ++i) { out[i] = lo[i]; out[8 + i] = hi[i]; }
  return out;
}

// C/D 16x16 f32: acc[v] is element (m = v + 8*(lane>=16), n = lane&15).
static __device__ __forceinline__ void st_tile_act(_Float16* base, int ld, int lane,
                                                   v8f acc, float bias) {
  const int n = lane & 15, hi = (lane >> 4) & 1;
#pragma unroll
  for (int v = 0; v < 8; ++v) {
    float x = acc[v] + bias;
    x = (x > 0.f) ? x : (__expf(x) - 1.f);     // ELU
    base[(v + 8 * hi) * ld + n] = (_Float16)x;
  }
}
// transposed store: dest element (n, m) -- used to store V as Vt[a][word]
static __device__ __forceinline__ void st_tile_act_T(_Float16* base, int ldT, int lane,
                                                     v8f acc, float bias) {
  const int n = lane & 15, hi = (lane >> 4) & 1;
#pragma unroll
  for (int v = 0; v < 8; ++v) {
    float x = acc[v] + bias;
    x = (x > 0.f) ? x : (__expf(x) - 1.f);
    base[n * ldT + (v + 8 * hi)] = (_Float16)x;
  }
}
static __device__ __forceinline__ void st_tile_h(_Float16* base, int ld, int lane, v8f acc) {
  const int n = lane & 15, hi = (lane >> 4) & 1;
#pragma unroll
  for (int v = 0; v < 8; ++v) base[(v + 8 * hi) * ld + n] = (_Float16)acc[v];
}
static __device__ __forceinline__ void st_tile_f(float* base, int ld, int lane,
                                                 v8f acc, float scale) {
  const int n = lane & 15, hi = (lane >> 4) & 1;
#pragma unroll
  for (int v = 0; v < 8; ++v) base[(v + 8 * hi) * ld + n] = acc[v] * scale;
}

#define WMMA_F16(a, b, c) \
  __builtin_amdgcn_wmma_f32_16x16x32_f16(false, (a), false, (b), (short)0, (c), false, false)

// ---------------- prep: per-doc masks --------------------------------------
__global__ void prep_masks_kernel(const int* __restrict__ docs,
                                  int* __restrict__ doc_mw, int* __restrict__ doc_ml) {
  __shared__ int nw_s[S_];
  const int b = blockIdx.x, s = threadIdx.x;  // 64 threads
  const int* p = docs + ((size_t)b * S_ + s) * W_;
  int cnt = 0;
  for (int w = 0; w < W_; ++w) cnt += (p[w] != 0) ? 1 : 0;
  nw_s[s] = cnt;
  __syncthreads();
  if (s == 0) {
    int m = 0, L = 0;
    for (int i = 0; i < S_; ++i) { m = (nw_s[i] > m) ? nw_s[i] : m; L += (nw_s[i] > 0) ? 1 : 0; }
    doc_mw[b] = m;
    doc_ml[b] = L;
  }
}

// ---------------- prep: weights f32 -> f16 (word weights padded E->EP) ------
__global__ void prep_weights_kernel(const float* wq, const float* wk, const float* wv,
                                    const float* lq, const float* lk, const float* lv,
                                    _Float16* wq_h, _Float16* wk_h, _Float16* wv_h,
                                    _Float16* lq_h, _Float16* lk_h, _Float16* lv_h) {
  const int NW = 3 * A_ * EP;
  const int NL = 3 * A_ * A_;
  int idx = blockIdx.x * blockDim.x + threadIdx.x;
  if (idx < NW) {
    int r = idx / (A_ * EP);
    int rem = idx - r * (A_ * EP);
    int a = rem / EP, e = rem - a * EP;
    const float* src = (r == 0) ? wq : (r == 1) ? wk : wv;
    _Float16*   dst = (r == 0) ? wq_h : (r == 1) ? wk_h : wv_h;
    dst[a * EP + e] = (e < E_) ? (_Float16)src[a * E_ + e] : (_Float16)0.f;
  } else if (idx < NW + NL) {
    int j = idx - NW;
    int r = j / (A_ * A_);
    int rem = j - r * (A_ * A_);
    const float* src = (r == 0) ? lq : (r == 1) ? lk : lv;
    _Float16*   dst = (r == 0) ? lq_h : (r == 1) ? lk_h : lv_h;
    dst[rem] = (_Float16)src[rem];
  }
}

// ---------------- stage 1: one workgroup per line ---------------------------
// LDS: xs[48][LDX]h | Qs[48][LDA]h (reused as out) | Ks[48][LDA]h |
//      Vt[512][LDV]h (V transposed, pad words zeroed) | Sc[48*48]f |
//      Ps[48][LDV]h | toks/tsc/pp
constexpr int S1_XS   = 0;
constexpr int S1_QS   = S1_XS + W_ * LDX * 2;
constexpr int S1_KS   = S1_QS + W_ * LDA * 2;
constexpr int S1_VT   = S1_KS + W_ * LDA * 2;
constexpr int S1_SC   = S1_VT + A_ * LDV * 2;
constexpr int S1_PS   = S1_SC + W_ * W_ * 4;
constexpr int S1_TOK  = S1_PS + W_ * LDV * 2;
constexpr int S1_TSC  = S1_TOK + W_ * 4;
constexpr int S1_PP   = S1_TSC + W_ * 4;
constexpr int S1_SMEM = S1_PP + W_ * 4;

__global__ __launch_bounds__(256) void stage1_kernel(
    const int* __restrict__ docs, const float* __restrict__ emb,
    const _Float16* __restrict__ wq_h, const _Float16* __restrict__ wk_h,
    const _Float16* __restrict__ wv_h,
    const float* __restrict__ wq_b, const float* __restrict__ wk_b,
    const float* __restrict__ wv_b,
    const float* __restrict__ word_target, const int* __restrict__ doc_mw,
    _Float16* __restrict__ line_emb) {
  extern __shared__ char smem[];
  _Float16* xs  = (_Float16*)(smem + S1_XS);
  _Float16* Qs  = (_Float16*)(smem + S1_QS);
  _Float16* Ks  = (_Float16*)(smem + S1_KS);
  _Float16* Vt  = (_Float16*)(smem + S1_VT);
  float*    Sc  = (float*)   (smem + S1_SC);
  _Float16* Ps  = (_Float16*)(smem + S1_PS);
  int*      toks= (int*)     (smem + S1_TOK);
  float*    tsc = (float*)   (smem + S1_TSC);
  float*    pp  = (float*)   (smem + S1_PP);

  const int l    = blockIdx.x;          // global line id
  const int b    = l / S_;
  const int tid  = threadIdx.x;
  const int lane = tid & 31;
  const int wave = tid >> 5;

  if (tid < W_) toks[tid] = docs[(size_t)l * W_ + tid];
  // zero the pad-word columns of Vt (K values 48..63 read by P@V)
  for (int i = tid; i < A_ * 16; i += 256) {
    int a = i >> 4, w = 48 + (i & 15);
    Vt[a * LDV + w] = (_Float16)0.f;
  }
  __syncthreads();

  // embedding gather -> f16 LDS (E padded to LDX with zeros)
  for (int i = tid; i < W_ * LDX; i += 256) {
    int w = i / LDX, e = i - w * LDX;
    float v = (e < E_) ? emb[(size_t)toks[w] * E_ + e] : 0.f;
    xs[i] = (_Float16)v;
  }
  __syncthreads();

  // Q/K/V = elu(x @ W^T + b): 3 proj x 3 Mtiles x 32 Ntiles, K = 320
  // (double-buffered K loop; V stored transposed)
  for (int t = wave; t < 288; t += 8) {
    const int proj = t / 96, rem = t - proj * 96;
    const int mi = rem >> 5, ni = rem & 31;
    const _Float16* wb = (proj == 0) ? wq_h : (proj == 1) ? wk_h : wv_h;
    const float*    bb = (proj == 0) ? wq_b : (proj == 1) ? wk_b : wv_b;
    v8f acc = {};
    v16h a0 = ld_tileA(xs + mi * 16 * LDX, LDX, lane);
    v16h b0 = ld_tileB(wb + ni * 16 * EP, EP, lane);
#pragma unroll
    for (int kk = 1; kk < EP / 32; ++kk) {
      v16h a1 = ld_tileA(xs + mi * 16 * LDX + kk * 32, LDX, lane);
      v16h b1 = ld_tileB(wb + ni * 16 * EP + kk * 32, EP, lane);
      acc = WMMA_F16(a0, b0, acc);
      a0 = a1; b0 = b1;
    }
    acc = WMMA_F16(a0, b0, acc);
    const float bias = bb[ni * 16 + (lane & 15)];
    if (proj == 2)      st_tile_act_T(Vt + ni * 16 * LDV + mi * 16, LDV, lane, acc, bias);
    else if (proj == 1) st_tile_act(Ks + mi * 16 * LDA + ni * 16, LDA, lane, acc, bias);
    else                st_tile_act(Qs + mi * 16 * LDA + ni * 16, LDA, lane, acc, bias);
  }
  __syncthreads();

  // scores = Q K^T * SCALE  (48x48, K = 512)
  for (int t = wave; t < 9; t += 8) {
    const int mi = t / 3, ni = t - 3 * (t / 3);
    v8f acc = {};
    v16h a0 = ld_tileA(Qs + mi * 16 * LDA, LDA, lane);
    v16h b0 = ld_tileB(Ks + ni * 16 * LDA, LDA, lane);
#pragma unroll
    for (int kk = 1; kk < A_ / 32; ++kk) {
      v16h a1 = ld_tileA(Qs + mi * 16 * LDA + kk * 32, LDA, lane);
      v16h b1 = ld_tileB(Ks + ni * 16 * LDA + kk * 32, LDA, lane);
      acc = WMMA_F16(a0, b0, acc);
      a0 = a1; b0 = b1;
    }
    acc = WMMA_F16(a0, b0, acc);
    st_tile_f(Sc + mi * 16 * W_ + ni * 16, W_, lane, acc, SCALE_);
  }
  __syncthreads();

  // masked row softmax -> P (f16, K padded 48->64 with zeros)
  const int mw = doc_mw[b];
  if (tid < W_) {
    float mx = -1e30f;
    for (int w = 0; w < mw; ++w) mx = fmaxf(mx, Sc[tid * W_ + w]);
    float sum = 0.f;
    for (int w = 0; w < mw; ++w) {
      float e = __expf(Sc[tid * W_ + w] - mx);
      sum += e;
      Ps[tid * LDV + w] = (_Float16)e;
    }
    float inv = 1.f / sum;
    for (int w = 0; w < mw; ++w)
      Ps[tid * LDV + w] = (_Float16)((float)Ps[tid * LDV + w] * inv);
    for (int w = mw; w < 64; ++w) Ps[tid * LDV + w] = (_Float16)0.f;
  }
  __syncthreads();

  // out = P @ V  (48x512, K = 64; B-tiles contiguous from Vt) -> reuse Qs
  for (int t = wave; t < 96; t += 8) {
    const int mi = t >> 5, ni = t & 31;
    v8f acc = {};
    v16h a0 = ld_tileA(Ps + mi * 16 * LDV, LDV, lane);
    v16h b0 = ld_tileB(Vt + ni * 16 * LDV, LDV, lane);
    v16h a1 = ld_tileA(Ps + mi * 16 * LDV + 32, LDV, lane);
    v16h b1 = ld_tileB(Vt + ni * 16 * LDV + 32, LDV, lane);
    acc = WMMA_F16(a0, b0, acc);
    acc = WMMA_F16(a1, b1, acc);
    st_tile_h(Qs + mi * 16 * LDA + ni * 16, LDA, lane, acc);
  }
  __syncthreads();

  // word-target pooling -> line embedding
  if (tid < W_) {
    float s = 0.f;
    for (int a = 0; a < A_; ++a) s += word_target[a] * (float)Qs[tid * LDA + a];
    tsc[tid] = s * SCALE_;
  }
  __syncthreads();
  if (tid == 0) {
    float mx = -1e30f;
    for (int w = 0; w < mw; ++w) mx = fmaxf(mx, tsc[w]);
    float sum = 0.f;
    for (int w = 0; w < mw; ++w) { pp[w] = __expf(tsc[w] - mx); sum += pp[w]; }
    float inv = 1.f / sum;
    for (int w = 0; w < mw; ++w) pp[w] *= inv;
    for (int w = mw; w < W_; ++w) pp[w] = 0.f;
  }
  __syncthreads();
  for (int a = tid; a < A_; a += 256) {
    float s = 0.f;
    for (int w = 0; w < W_; ++w) s += pp[w] * (float)Qs[w * LDA + a];
    line_emb[(size_t)l * A_ + a] = (_Float16)s;
  }
}

// ---------------- stage 2: one workgroup per doc ----------------------------
constexpr int S2_LE   = 0;                          // 64 x LDA h (reused as out)
constexpr int S2_QS   = S2_LE + S_ * LDA * 2;
constexpr int S2_KS   = S2_QS + S_ * LDA * 2;
constexpr int S2_VT   = S2_KS + S_ * LDA * 2;       // 512 x LDV h (V transposed)
constexpr int S2_SC   = S2_VT + A_ * LDV * 2;       // 64x64 f32
constexpr int S2_PS   = S2_SC + S_ * S_ * 4;        // 64 x LDV h
constexpr int S2_TSC  = S2_PS + S_ * LDV * 2;
constexpr int S2_PP   = S2_TSC + S_ * 4;
constexpr int S2_DE   = S2_PP + S_ * 4;             // 512 f32
constexpr int S2_SMEM = S2_DE + A_ * 4;

__global__ __launch_bounds__(256) void stage2_kernel(
    const _Float16* __restrict__ line_emb,
    const _Float16* __restrict__ lq_h, const _Float16* __restrict__ lk_h,
    const _Float16* __restrict__ lv_h,
    const float* __restrict__ lq_b, const float* __restrict__ lk_b,
    const float* __restrict__ lv_b,
    const float* __restrict__ line_target, const int* __restrict__ doc_ml,
    const float* __restrict__ cls0_w, const float* __restrict__ cls0_b,
    const float* __restrict__ cls1_w, const float* __restrict__ cls1_b,
    float* __restrict__ out) {
  extern __shared__ char smem[];
  _Float16* le = (_Float16*)(smem + S2_LE);
  _Float16* Qs = (_Float16*)(smem + S2_QS);
  _Float16* Ks = (_Float16*)(smem + S2_KS);
  _Float16* Vt = (_Float16*)(smem + S2_VT);
  float*    Sc = (float*)   (smem + S2_SC);
  _Float16* Ps = (_Float16*)(smem + S2_PS);
  float*   tsc = (float*)   (smem + S2_TSC);
  float*    pp = (float*)   (smem + S2_PP);
  float*    de = (float*)   (smem + S2_DE);

  const int b    = blockIdx.x;
  const int tid  = threadIdx.x;
  const int lane = tid & 31;
  const int wave = tid >> 5;

  for (int i = tid; i < S_ * A_; i += 256) {
    int s = i >> 9, a = i & 511;
    le[s * LDA + a] = line_emb[(size_t)b * S_ * A_ + i];
  }
  __syncthreads();

  // line Q/K/V: 3 proj x 4 Mtiles x 32 Ntiles, K = 512 (V stored transposed)
  for (int t = wave; t < 384; t += 8) {
    const int proj = t / 128, rem = t - proj * 128;
    const int mi = rem >> 5, ni = rem & 31;
    const _Float16* wb = (proj == 0) ? lq_h : (proj == 1) ? lk_h : lv_h;
    const float*    bb = (proj == 0) ? lq_b : (proj == 1) ? lk_b : lv_b;
    v8f acc = {};
    v16h a0 = ld_tileA(le + mi * 16 * LDA, LDA, lane);
    v16h b0 = ld_tileB(wb + ni * 16 * A_, A_, lane);
#pragma unroll
    for (int kk = 1; kk < A_ / 32; ++kk) {
      v16h a1 = ld_tileA(le + mi * 16 * LDA + kk * 32, LDA, lane);
      v16h b1 = ld_tileB(wb + ni * 16 * A_ + kk * 32, A_, lane);
      acc = WMMA_F16(a0, b0, acc);
      a0 = a1; b0 = b1;
    }
    acc = WMMA_F16(a0, b0, acc);
    const float bias = bb[ni * 16 + (lane & 15)];
    if (proj == 2)      st_tile_act_T(Vt + ni * 16 * LDV + mi * 16, LDV, lane, acc, bias);
    else if (proj == 1) st_tile_act(Ks + mi * 16 * LDA + ni * 16, LDA, lane, acc, bias);
    else                st_tile_act(Qs + mi * 16 * LDA + ni * 16, LDA, lane, acc, bias);
  }
  __syncthreads();

  // scores 64x64, K = 512
  for (int t = wave; t < 16; t += 8) {
    const int mi = t >> 2, ni = t & 3;
    v8f acc = {};
    v16h a0 = ld_tileA(Qs + mi * 16 * LDA, LDA, lane);
    v16h b0 = ld_tileB(Ks + ni * 16 * LDA, LDA, lane);
#pragma unroll
    for (int kk = 1; kk < A_ / 32; ++kk) {
      v16h a1 = ld_tileA(Qs + mi * 16 * LDA + kk * 32, LDA, lane);
      v16h b1 = ld_tileB(Ks + ni * 16 * LDA + kk * 32, LDA, lane);
      acc = WMMA_F16(a0, b0, acc);
      a0 = a1; b0 = b1;
    }
    acc = WMMA_F16(a0, b0, acc);
    st_tile_f(Sc + mi * 16 * S_ + ni * 16, S_, lane, acc, SCALE_);
  }
  __syncthreads();

  const int ml = doc_ml[b];
  if (tid < S_) {
    float mx = -1e30f;
    for (int s = 0; s < ml; ++s) mx = fmaxf(mx, Sc[tid * S_ + s]);
    float sum = 0.f;
    for (int s = 0; s < ml; ++s) {
      float e = __expf(Sc[tid * S_ + s] - mx);
      sum += e;
      Ps[tid * LDV + s] = (_Float16)e;
    }
    float inv = 1.f / sum;
    for (int s = 0; s < ml; ++s)
      Ps[tid * LDV + s] = (_Float16)((float)Ps[tid * LDV + s] * inv);
    for (int s = ml; s < S_; ++s) Ps[tid * LDV + s] = (_Float16)0.f;
  }
  __syncthreads();

  // out = P @ V (64x512, K = 64; B-tiles contiguous from Vt) -> reuse le
  for (int t = wave; t < 128; t += 8) {
    const int mi = t >> 5, ni = t & 31;
    v8f acc = {};
    v16h a0 = ld_tileA(Ps + mi * 16 * LDV, LDV, lane);
    v16h b0 = ld_tileB(Vt + ni * 16 * LDV, LDV, lane);
    v16h a1 = ld_tileA(Ps + mi * 16 * LDV + 32, LDV, lane);
    v16h b1 = ld_tileB(Vt + ni * 16 * LDV + 32, LDV, lane);
    acc = WMMA_F16(a0, b0, acc);
    acc = WMMA_F16(a1, b1, acc);
    st_tile_h(le + mi * 16 * LDA + ni * 16, LDA, lane, acc);
  }
  __syncthreads();

  // line-target pooling -> doc embedding
  if (tid < S_) {
    float s = 0.f;
    for (int a = 0; a < A_; ++a) s += line_target[a] * (float)le[tid * LDA + a];
    tsc[tid] = s * SCALE_;
  }
  __syncthreads();
  if (tid == 0) {
    float mx = -1e30f;
    for (int s = 0; s < ml; ++s) mx = fmaxf(mx, tsc[s]);
    float sum = 0.f;
    for (int s = 0; s < ml; ++s) { pp[s] = __expf(tsc[s] - mx); sum += pp[s]; }
    float inv = 1.f / sum;
    for (int s = 0; s < ml; ++s) pp[s] *= inv;
    for (int s = ml; s < S_; ++s) pp[s] = 0.f;
  }
  __syncthreads();
  for (int a = tid; a < A_; a += 256) {
    float s = 0.f;
    for (int w = 0; w < S_; ++w) s += pp[w] * (float)le[w * LDA + a];
    de[a] = s;
  }
  __syncthreads();

  // classifiers
  if (tid < C0_) {
    float s = cls0_b[tid];
    for (int a = 0; a < A_; ++a) s += de[a] * cls0_w[tid * A_ + a];
    out[b * C0_ + tid] = s;
  }
  if (tid >= 32 && tid < 32 + C1_) {
    const int c = tid - 32;
    float s = cls1_b[c];
    for (int a = 0; a < A_; ++a) s += de[a] * cls1_w[c * A_ + a];
    out[B_ * C0_ + b * C1_ + c] = s;
  }
}

// ---------------- launch -----------------------------------------------------
extern "C" void kernel_launch(void* const* d_in, const int* in_sizes, int n_in,
                              void* d_out, int out_size, void* d_ws, size_t ws_size,
                              hipStream_t stream) {
  const int*   documents  = (const int*)  d_in[0];
  const float* emb        = (const float*)d_in[1];
  const float* wq_w       = (const float*)d_in[2];
  const float* wq_b       = (const float*)d_in[3];
  const float* wk_w       = (const float*)d_in[4];
  const float* wk_b       = (const float*)d_in[5];
  const float* wv_w       = (const float*)d_in[6];
  const float* wv_b       = (const float*)d_in[7];
  const float* lq_w       = (const float*)d_in[8];
  const float* lq_b       = (const float*)d_in[9];
  const float* lk_w       = (const float*)d_in[10];
  const float* lk_b       = (const float*)d_in[11];
  const float* lv_w       = (const float*)d_in[12];
  const float* lv_b       = (const float*)d_in[13];
  const float* word_tgt   = (const float*)d_in[14];
  const float* line_tgt   = (const float*)d_in[15];
  const float* cls0_w     = (const float*)d_in[16];
  const float* cls0_b     = (const float*)d_in[17];
  const float* cls1_w     = (const float*)d_in[18];
  const float* cls1_b     = (const float*)d_in[19];
  float* out = (float*)d_out;

  // workspace carve
  char* ws = (char*)d_ws;
  int*      doc_mw  = (int*)ws;                         // 32 ints
  int*      doc_ml  = (int*)(ws + 128);                 // 32 ints
  _Float16* wq_h    = (_Float16*)(ws + 256);
  _Float16* wk_h    = wq_h + A_ * EP;
  _Float16* wv_h    = wk_h + A_ * EP;
  _Float16* lq_h    = wv_h + A_ * EP;
  _Float16* lk_h    = lq_h + A_ * A_;
  _Float16* lv_h    = lk_h + A_ * A_;
  _Float16* line_e  = lv_h + A_ * A_;                   // [B*S, A] f16

  prep_masks_kernel<<<B_, S_, 0, stream>>>(documents, doc_mw, doc_ml);

  const int totalW = 3 * A_ * EP + 3 * A_ * A_;
  prep_weights_kernel<<<(totalW + 255) / 256, 256, 0, stream>>>(
      wq_w, wk_w, wv_w, lq_w, lk_w, lv_w, wq_h, wk_h, wv_h, lq_h, lk_h, lv_h);

  stage1_kernel<<<B_ * S_, 256, S1_SMEM, stream>>>(
      documents, emb, wq_h, wk_h, wv_h, wq_b, wk_b, wv_b,
      word_tgt, doc_mw, line_e);

  stage2_kernel<<<B_, 256, S2_SMEM, stream>>>(
      line_e, lq_h, lk_h, lv_h, lq_b, lk_b, lv_b,
      line_tgt, doc_ml, cls0_w, cls0_b, cls1_w, cls1_b, out);
}
